// AspectAttention_66082366816536
// MI455X (gfx1250) — compile-verified, matrix operand order checked by reading, not run on hardware
//
#include <hip/hip_runtime.h>

// ---------------------------------------------------------------------------
// AspectAttention for MI455X (gfx1250, wave32, WMMA).
//   out[b,n] = sum_d softmax_n(tanh(h@W + b_a))[n,d] * h[b,n,d]
// GEMM via v_wmma_f32_16x16x32_bf16 with a hi/lo bf16 split of both operands
// (3 WMMAs per K-step) for ~fp32 accuracy at bf16 matrix rates.
// LDS: bf16 staging buffers and the f32 score buffer are overlaid (they are
// live in disjoint phases) -> ~68 KB -> 2 workgroups per CU in CU mode.
// ---------------------------------------------------------------------------

typedef __attribute__((ext_vector_type(16))) __bf16       v16bf;
typedef __attribute__((ext_vector_type(8)))  float        v8f;
typedef __attribute__((ext_vector_type(4)))  unsigned int u32x4;
typedef __attribute__((ext_vector_type(4)))  float        f32x4;

union FragU { u32x4 u[2]; v16bf v; };

#define NBLK        64
#define DMODEL      256
#define KTILES      8        // 256 / 32
#define LDS_SSTR    260      // f32 row stride (pad: conflict-free column walks)
#define LDS_HSTR    264      // bf16 row stride (pad: rows hit distinct bank groups)
#define SMEM_BYTES  (NBLK * LDS_HSTR * 2 * 2)   // 67584 B >= NBLK*LDS_SSTR*4

// Fast, safe tanh for the softmax path: e = exp(-2|x|) in (0,1] never
// overflows; (1-e)/(1+e) = tanh(|x|); restore sign. ~7 VALU ops via the
// v_exp_f32 / v_rcp_f32 transcendental pipe vs ~20 for ocml tanhf.
__device__ __forceinline__ float fast_tanhf(float x) {
  float ax = __builtin_fabsf(x);
  float e  = __expf(-2.0f * ax);
  float r  = (1.0f - e) * __builtin_amdgcn_rcpf(1.0f + e);
  return __builtin_copysignf(r, x);
}

// ---------------------------------------------------------------------------
// Pre-pass: W_a (f32 [K=256][N=256]) -> bf16 hi/lo, pre-swizzled into the
// WMMA B-fragment layout (ISA 7.12.2): for tile (kt, c), lane l holds
// column n = c*16 + (l&15); its 16 values are K = kt*32 + (l>=16 ? 8:0) +
// {0..7, 16..23}. Stored contiguously (16 bf16 per lane) so the main kernel's
// B loads are two coalesced b128 global loads per fragment.
// ---------------------------------------------------------------------------
__global__ void wa_swizzle_kernel(const float* __restrict__ W,
                                  __bf16* __restrict__ bhi,
                                  __bf16* __restrict__ blo) {
  unsigned t    = blockIdx.x * blockDim.x + threadIdx.x;  // 0..65535
  unsigned i    = t & 15;
  unsigned lane = (t >> 4) & 31;
  unsigned c    = (t >> 9) & 15;
  unsigned kt   = t >> 13;

  unsigned k = kt * 32 + ((lane >> 4) ? 8u : 0u) + (i < 8 ? i : i + 8);
  unsigned n = c * 16 + (lane & 15);

  float  x  = W[k * DMODEL + n];
  __bf16 hi = (__bf16)x;
  bhi[t] = hi;
  blo[t] = (__bf16)(x - (float)hi);
}

// ---------------------------------------------------------------------------
// Main kernel: one workgroup (256 threads = 8 waves) per batch element.
// ---------------------------------------------------------------------------
__global__ __launch_bounds__(256) void aspect_attn_kernel(
    const float*  __restrict__ h,
    const float*  __restrict__ b_a,
    const __bf16* __restrict__ wBhi,
    const __bf16* __restrict__ wBlo,
    float*        __restrict__ out) {
  // Phase 1 (staging + GEMM): smem = hHi[64*264] ++ hLo[64*264]  (bf16)
  // Phase 2 (scores + softmax): smem = sS[64*260]                (f32)
  __shared__ __align__(16) char smem[SMEM_BYTES];
  __bf16* hHi = (__bf16*)smem;
  __bf16* hLo = hHi + NBLK * LDS_HSTR;
  float*  sS  = (float*)smem;

  const unsigned tid  = threadIdx.x;
  const unsigned lane = tid & 31;
  const unsigned wv   = tid >> 5;
  const unsigned b    = blockIdx.x;
  const float* hb = h + (size_t)b * (NBLK * DMODEL);

  // ---- Stage h_b -> bf16 hi/lo in LDS (coalesced float4 loads) -----------
  for (int j = 0; j < 16; ++j) {
    unsigned fidx = tid + j * 256;          // float4 index 0..4095
    unsigned row  = fidx >> 6;              // / (256/4)
    unsigned col  = (fidx & 63) * 4;
    f32x4 x = *(const f32x4*)(hb + (size_t)fidx * 4);
    unsigned base = row * LDS_HSTR + col;
#pragma unroll
    for (int e = 0; e < 4; ++e) {
      float  v  = x[e];
      __bf16 hi = (__bf16)v;
      hHi[base + e] = hi;
      hLo[base + e] = (__bf16)(v - (float)hi);
    }
  }
  __syncthreads();

  // ---- GEMM: wave w owns rows [rBase, rBase+16), col-tiles cgBase..+7 ----
  const unsigned rBase  = (wv & 3) * 16;
  const unsigned cgBase = (wv >> 2) * 8;
  const unsigned rowA   = rBase + (lane & 15);
  const unsigned colSel = (lane >> 4) * 8;   // A-fragment K-offset per lane half

  v8f acc[8];
  const v8f vzero = {0.f, 0.f, 0.f, 0.f, 0.f, 0.f, 0.f, 0.f};
#pragma unroll
  for (int t = 0; t < 8; ++t) acc[t] = vzero;

  for (int kt = 0; kt < KTILES; ++kt) {
    // A fragments (hi, lo) from LDS: two b128 each (K {0..7} and {16..23}).
    FragU aHi, aLo;
    const __bf16* pHi = &hHi[rowA * LDS_HSTR + kt * 32 + colSel];
    const __bf16* pLo = &hLo[rowA * LDS_HSTR + kt * 32 + colSel];
    aHi.u[0] = *(const u32x4*)(pHi);
    aHi.u[1] = *(const u32x4*)(pHi + 16);
    aLo.u[0] = *(const u32x4*)(pLo);
    aLo.u[1] = *(const u32x4*)(pLo + 16);

#pragma unroll
    for (int t = 0; t < 8; ++t) {
      unsigned cg = cgBase + t;
      const __bf16* pBh = wBhi + (size_t)(((kt * 16 + cg) * 32 + lane) * 16);
      const __bf16* pBl = wBlo + (size_t)(((kt * 16 + cg) * 32 + lane) * 16);
      FragU bHi, bLo;
      bHi.u[0] = *(const u32x4*)(pBh);
      bHi.u[1] = *(const u32x4*)(pBh + 8);
      bLo.u[0] = *(const u32x4*)(pBl);
      bLo.u[1] = *(const u32x4*)(pBl + 8);

      // Split-precision product: hi*hi + hi*lo + lo*hi (lo*lo negligible).
      acc[t] = __builtin_amdgcn_wmma_f32_16x16x32_bf16(
          false, aHi.v, false, bHi.v, (short)0, acc[t], false, false);
      acc[t] = __builtin_amdgcn_wmma_f32_16x16x32_bf16(
          false, aHi.v, false, bLo.v, (short)0, acc[t], false, false);
      acc[t] = __builtin_amdgcn_wmma_f32_16x16x32_bf16(
          false, aLo.v, false, bHi.v, (short)0, acc[t], false, false);
    }
  }

  // All waves are done reading hHi/hLo; the same LDS now becomes sS.
  __syncthreads();

  // ---- bias + tanh, scatter C fragments into padded f32 LDS --------------
  {
    const unsigned nCol   = lane & 15;
    const unsigned rowOff = (lane >> 4) * 8;   // C layout: lanes 16-31 are M+8
#pragma unroll
    for (int t = 0; t < 8; ++t) {
      unsigned col = (cgBase + t) * 16 + nCol;
#pragma unroll
      for (int v = 0; v < 8; ++v) {
        unsigned row = rBase + rowOff + v;
        float s = acc[t][v] + b_a[row * DMODEL + col];
        sS[row * LDS_SSTR + col] = fast_tanhf(s);
      }
    }
  }
  __syncthreads();

  // ---- softmax over the n_block axis: one column d per thread ------------
  {
    const unsigned d = tid;                 // 0..255
    float m = -3.402823466e38f;
#pragma unroll 8
    for (int n = 0; n < NBLK; ++n) m = fmaxf(m, sS[n * LDS_SSTR + d]);
    float sum = 0.f;
#pragma unroll 8
    for (int n = 0; n < NBLK; ++n) sum += __expf(sS[n * LDS_SSTR + d] - m);
    float inv = __builtin_amdgcn_rcpf(sum);  // sum >= 1 (max term included)
#pragma unroll 8
    for (int n = 0; n < NBLK; ++n)
      sS[n * LDS_SSTR + d] = __expf(sS[n * LDS_SSTR + d] - m) * inv;
  }
  __syncthreads();

  // ---- out[b,n] = sum_d a[n,d] * h[b,n,d]  (h re-read; hot in L2) --------
  for (int rr = 0; rr < 8; ++rr) {
    unsigned n = wv * 8 + rr;
    const float* hrow = hb + (size_t)n * DMODEL;
    float a = 0.f;
#pragma unroll
    for (int j = 0; j < 8; ++j) {
      unsigned d = lane + j * 32;
      a += sS[n * LDS_SSTR + d] * hrow[d];
    }
#pragma unroll
    for (int off = 16; off > 0; off >>= 1)
      a += __shfl_down(a, off, 32);         // wave32 reduction
    if (lane == 0) out[(size_t)b * NBLK + n] = a;
  }
}

// ---------------------------------------------------------------------------
extern "C" void kernel_launch(void* const* d_in, const int* in_sizes, int n_in,
                              void* d_out, int out_size, void* d_ws, size_t ws_size,
                              hipStream_t stream) {
  (void)in_sizes; (void)n_in; (void)out_size; (void)ws_size;
  const float* h   = (const float*)d_in[0];
  const float* W_a = (const float*)d_in[1];
  const float* b_a = (const float*)d_in[2];
  float* out = (float*)d_out;

  // d_ws: [0, 128KB) = W hi fragments, [128KB, 256KB) = W lo fragments.
  __bf16* wBhi = (__bf16*)d_ws;
  __bf16* wBlo = (__bf16*)((char*)d_ws + (size_t)DMODEL * DMODEL * sizeof(__bf16));

  wa_swizzle_kernel<<<256, 256, 0, stream>>>(W_a, wBhi, wBlo);
  aspect_attn_kernel<<<4096, 256, 0, stream>>>(h, b_a, wBhi, wBlo, out);
}
